// AggrHGraphConvLayer_79285096284406
// MI455X (gfx1250) — matrix operand order repaired and checked by reading, not compiled, not run on hardware
//
#include <hip/hip_runtime.h>

#define SVC_N  20000
#define INST_N 100000
#define NODE_N 10000
#define HID    128
#define E_SC   640000
#define E_IN   200000
#define E_NI   200000

typedef __attribute__((ext_vector_type(16))) __bf16 v16bf;
typedef __attribute__((ext_vector_type(8)))  float  v8f;

// ---------------- utility kernels ----------------

__global__ void zero_f32(float* __restrict__ p, int n) {
  int t = blockIdx.x * blockDim.x + threadIdx.x;
  if (t < n) p[t] = 0.0f;
}

__global__ void deg_count(const int* __restrict__ idx, float* __restrict__ deg, int n) {
  int t = blockIdx.x * blockDim.x + threadIdx.x;
  if (t < n) atomicAdd(&deg[idx[t]], 1.0f);
}

// Pre-swizzle W[128][128] (row-major fp32) into WMMA-B bf16 hi/lo layout:
// Bp[((kb*8 + nt)*32 + lane)*16 + i]  holds  W[kb*32 + (lane>>4)*16 + i][nt*16 + (lane&15)]
__global__ void pack_w(const float* __restrict__ W,
                       __bf16* __restrict__ Bhi, __bf16* __restrict__ Blo) {
  int t = blockIdx.x * blockDim.x + threadIdx.x;
  if (t >= HID * HID) return;
  int i    = t & 15;
  int lane = (t >> 4) & 31;
  int ntkb = t >> 9;          // = kb*8 + nt
  int kb = ntkb >> 3;
  int nt = ntkb & 7;
  int g  = lane >> 4;
  int n  = (lane & 15) + nt * 16;
  int k  = kb * 32 + g * 16 + i;
  float w = W[k * HID + n];
  __bf16 hi = (__bf16)w;
  Bhi[t] = hi;
  Blo[t] = (__bf16)(w - (float)hi);
}

// ---------------- WMMA GEMM: H = (X * rsqrt(deg_src)) @ W ----------------
// One wave computes rows [row0, row0+16) x all 128 cols, bf16x3 split precision.

__global__ __launch_bounds__(256) void gemm_bf16x3(
    const float* __restrict__ X, const float* __restrict__ deg,
    const __bf16* __restrict__ Bhi, const __bf16* __restrict__ Blo,
    float* __restrict__ H, int n_rows)
{
  int lane = threadIdx.x & 31;
  int wave = threadIdx.x >> 5;
  int row0 = (blockIdx.x * 8 + wave) * 16;
  if (row0 >= n_rows) return;          // wave-uniform skip, EXEC stays all-ones
  int g  = lane >> 4;
  int ml = lane & 15;
  int arow = row0 + ml;
  float s = rsqrtf(fmaxf(deg[arow], 1.0f));
  const float* ap = X + (size_t)arow * HID;

  v8f c[8];
  #pragma unroll
  for (int nt = 0; nt < 8; nt++)
    #pragma unroll
    for (int r = 0; r < 8; r++) c[nt][r] = 0.0f;

  #pragma unroll
  for (int kb = 0; kb < 4; kb++) {
    int k0 = kb * 32;
    // A halves i=0..15 -> K = k0 + g*16 + i
    const float4* q0 = (const float4*)(ap + k0 + g * 16);
    const float4* q1 = (const float4*)(ap + k0 + g * 16 + 8);
    float4 A0 = q0[0], A1 = q0[1], A2 = q1[0], A3 = q1[1];
    float fv[16] = {A0.x, A0.y, A0.z, A0.w, A1.x, A1.y, A1.z, A1.w,
                    A2.x, A2.y, A2.z, A2.w, A3.x, A3.y, A3.z, A3.w};
    v16bf ah, al;
    #pragma unroll
    for (int i = 0; i < 16; i++) {
      float f  = fv[i] * s;
      __bf16 hi = (__bf16)f;
      ah[i] = hi;
      al[i] = (__bf16)(f - (float)hi);
    }
    #pragma unroll
    for (int nt = 0; nt < 8; nt++) {
      size_t off = (size_t)((kb * 8 + nt) * 32 + lane) * 16;
      v16bf bh = *(const v16bf*)(Bhi + off);
      v16bf bl = *(const v16bf*)(Blo + off);
      c[nt] = __builtin_amdgcn_wmma_f32_16x16x32_bf16(false, ah, false, bh, (short)0, c[nt], false, false);
      c[nt] = __builtin_amdgcn_wmma_f32_16x16x32_bf16(false, ah, false, bl, (short)0, c[nt], false, false);
      c[nt] = __builtin_amdgcn_wmma_f32_16x16x32_bf16(false, al, false, bh, (short)0, c[nt], false, false);
    }
  }

  #pragma unroll
  for (int nt = 0; nt < 8; nt++) {
    #pragma unroll
    for (int r = 0; r < 8; r++) {
      int row = row0 + g * 8 + r;
      int col = nt * 16 + ml;
      H[(size_t)row * HID + col] = c[nt][r];
    }
  }
}

// ---------------- edge scatter: out[dst] += H[src] ----------------
// One wave per edge; each lane handles 4 of 128 features.

__global__ __launch_bounds__(256) void scatter_add(
    const float* __restrict__ H, const int* __restrict__ src,
    const int* __restrict__ dst, float* __restrict__ out, int n_edges)
{
  int t = blockIdx.x * blockDim.x + threadIdx.x;
  int e = t >> 5;
  if (e >= n_edges) return;
  int lane = t & 31;
  const float4 v = *(const float4*)(H + (size_t)src[e] * HID + lane * 4);
  float* o = out + (size_t)dst[e] * HID + lane * 4;
  atomicAdd(o + 0, v.x);
  atomicAdd(o + 1, v.y);
  atomicAdd(o + 2, v.z);
  atomicAdd(o + 3, v.w);
}

// ---------------- epilogue: x = leaky(x * rsqrt(deg_dst) + b) ----------------

__global__ void finalize(float* __restrict__ out, const float* __restrict__ deg,
                         const float* __restrict__ b, int n_nodes) {
  int t = blockIdx.x * blockDim.x + threadIdx.x;
  if (t >= n_nodes * HID) return;
  int node = t >> 7;
  int col  = t & 127;
  float s = rsqrtf(fmaxf(deg[node], 1.0f));
  float x = out[t] * s + b[col];
  out[t] = (x >= 0.0f) ? x : 0.01f * x;
}

// ---------------- launch ----------------

extern "C" void kernel_launch(void* const* d_in, const int* in_sizes, int n_in,
                              void* d_out, int out_size, void* d_ws, size_t ws_size,
                              hipStream_t stream) {
  (void)in_sizes; (void)n_in; (void)out_size; (void)ws_size;
  const float* svc_feat  = (const float*)d_in[0];
  const float* inst_feat = (const float*)d_in[1];
  const float* node_feat = (const float*)d_in[2];
  const float* W_svc  = (const float*)d_in[3];
  const float* b_svc  = (const float*)d_in[4];
  const float* W_inst = (const float*)d_in[5];
  const float* b_inst = (const float*)d_in[6];
  const float* W_node = (const float*)d_in[7];
  const float* b_node = (const float*)d_in[8];
  const int* sc_src = (const int*)d_in[9];
  const int* sc_dst = (const int*)d_in[10];
  const int* in_src = (const int*)d_in[11];
  const int* in_dst = (const int*)d_in[12];
  const int* ni_src = (const int*)d_in[13];
  const int* ni_dst = (const int*)d_in[14];

  float* out      = (float*)d_out;
  float* out_node = out;                                  // [NODE_N,128]
  float* out_inst = out + (size_t)NODE_N * HID;           // [INST_N,128]
  float* out_svc  = out + (size_t)(NODE_N + INST_N) * HID;// [SVC_N,128]

  char* ws = (char*)d_ws;
  float* h_svc  = (float*)ws; ws += (size_t)SVC_N  * HID * 4;
  float* h_inst = (float*)ws; ws += (size_t)INST_N * HID * 4;
  float* h_node = (float*)ws; ws += (size_t)NODE_N * HID * 4;
  float* degs = (float*)ws;
  float* deg_sc_s = degs;
  float* deg_sc_d = deg_sc_s + SVC_N;
  float* deg_in_s = deg_sc_d + SVC_N;
  float* deg_in_d = deg_in_s + INST_N;
  float* deg_ni_s = deg_in_d + NODE_N;
  float* deg_ni_d = deg_ni_s + NODE_N;
  ws += (size_t)(2 * SVC_N + 2 * INST_N + 2 * NODE_N) * 4;
  __bf16* p = (__bf16*)ws;
  __bf16* Whi_svc  = p; p += HID * HID;
  __bf16* Wlo_svc  = p; p += HID * HID;
  __bf16* Whi_inst = p; p += HID * HID;
  __bf16* Wlo_inst = p; p += HID * HID;
  __bf16* Whi_node = p; p += HID * HID;
  __bf16* Wlo_node = p; p += HID * HID;

  const int TB = 256;

  // zero accumulation targets (harness poisons buffers)
  int n_out = (NODE_N + INST_N + SVC_N) * HID;
  zero_f32<<<(n_out + TB - 1) / TB, TB, 0, stream>>>(out, n_out);
  int n_deg = 2 * SVC_N + 2 * INST_N + 2 * NODE_N;
  zero_f32<<<(n_deg + TB - 1) / TB, TB, 0, stream>>>(degs, n_deg);

  // degrees
  deg_count<<<(E_SC + TB - 1) / TB, TB, 0, stream>>>(sc_src, deg_sc_s, E_SC);
  deg_count<<<(E_SC + TB - 1) / TB, TB, 0, stream>>>(sc_dst, deg_sc_d, E_SC);
  deg_count<<<(E_IN + TB - 1) / TB, TB, 0, stream>>>(in_src, deg_in_s, E_IN);
  deg_count<<<(E_IN + TB - 1) / TB, TB, 0, stream>>>(in_dst, deg_in_d, E_IN);
  deg_count<<<(E_NI + TB - 1) / TB, TB, 0, stream>>>(ni_src, deg_ni_s, E_NI);
  deg_count<<<(E_NI + TB - 1) / TB, TB, 0, stream>>>(ni_dst, deg_ni_d, E_NI);

  // weight pre-swizzle (bf16 hi/lo, WMMA-B layout)
  int n_w = HID * HID;
  pack_w<<<(n_w + TB - 1) / TB, TB, 0, stream>>>(W_svc,  Whi_svc,  Wlo_svc);
  pack_w<<<(n_w + TB - 1) / TB, TB, 0, stream>>>(W_inst, Whi_inst, Wlo_inst);
  pack_w<<<(n_w + TB - 1) / TB, TB, 0, stream>>>(W_node, Whi_node, Wlo_node);

  // GEMMs: H = (feat * rsqrt(deg_src)) @ W
  gemm_bf16x3<<<(SVC_N  + 127) / 128, 256, 0, stream>>>(svc_feat,  deg_sc_s, Whi_svc,  Wlo_svc,  h_svc,  SVC_N);
  gemm_bf16x3<<<(INST_N + 127) / 128, 256, 0, stream>>>(inst_feat, deg_in_s, Whi_inst, Wlo_inst, h_inst, INST_N);
  gemm_bf16x3<<<(NODE_N + 127) / 128, 256, 0, stream>>>(node_feat, deg_ni_s, Whi_node, Wlo_node, h_node, NODE_N);

  // edge aggregation straight into output regions
  scatter_add<<<((size_t)E_SC * 32 + TB - 1) / TB, TB, 0, stream>>>(h_svc,  sc_src, sc_dst, out_svc,  E_SC);
  scatter_add<<<((size_t)E_IN * 32 + TB - 1) / TB, TB, 0, stream>>>(h_inst, in_src, in_dst, out_node, E_IN);
  scatter_add<<<((size_t)E_NI * 32 + TB - 1) / TB, TB, 0, stream>>>(h_node, ni_src, ni_dst, out_inst, E_NI);

  // epilogue: normalize by dst degree, add bias, leaky relu (in place)
  finalize<<<(NODE_N * HID + TB - 1) / TB, TB, 0, stream>>>(out_node, deg_in_d, b_inst, NODE_N);
  finalize<<<(INST_N * HID + TB - 1) / TB, TB, 0, stream>>>(out_inst, deg_ni_d, b_node, INST_N);
  finalize<<<(SVC_N  * HID + TB - 1) / TB, TB, 0, stream>>>(out_svc,  deg_sc_d, b_svc,  SVC_N);
}
